// CylindricalPointEncoder_45861660786968
// MI455X (gfx1250) — matrix operand order repaired and checked by reading, not compiled, not run on hardware
//
#include <hip/hip_runtime.h>
#include <hip/hip_bf16.h>
#include <cstdint>
#include <cstddef>

// ---------------------------------------------------------------------------
// CDNA5 (gfx1250) fused voxel-feature-encoder pipeline.
//   stage 0: channel stats of feats (9ch)
//   stage 1: fold BN into (s,t);  y1 = BN0(x) @ W1 + b1  (scalar, K=9)
//   stage 2: y2 = relu(BN1(y1)) @ W2 + b2   [WMMA f16 16x16x32, TDM W-staging]
//   stage 3: y3 = relu(BN2(y2)) @ W3 + b3   [WMMA]
//   stage 4: relu(BN3(y3)) @ W4 + b4 -> fused segment-max scatter (atomicMax
//            on order-preserving uint encoding; 32MB target lives in L2)
//   stage 5: decode pooled in place + emit sorted unique coords (int64)
// All GEMMs accumulate per-channel sum/sumsq for the *next* layer's BN.
// ---------------------------------------------------------------------------

typedef _Float16 h16;
typedef __attribute__((ext_vector_type(16))) _Float16 v16h;
typedef __attribute__((ext_vector_type(8)))  float    v8f;
typedef unsigned int v4u __attribute__((ext_vector_type(4)));
typedef int          v4i __attribute__((ext_vector_type(4)));
typedef int          v8i __attribute__((ext_vector_type(8)));

#ifndef __has_builtin
#define __has_builtin(x) 0
#endif

#if __has_builtin(__builtin_amdgcn_tensor_load_to_lds) && \
    __has_builtin(__builtin_amdgcn_s_wait_tensorcnt)
#define VFE_HAS_TDM 1
#if __has_include(<hip/amd_detail/amd_gfx1250_TDM.h>)
#define VFE_TDM6 1   // therock headers present -> 6-arg builtin (clang-23)
#else
#define VFE_TDM6 0   // ROCm 7.2 -> 5-arg builtin (clang-22)
#endif
#else
#define VFE_HAS_TDM 0
#endif

static __device__ __forceinline__ int imin(int a, int b) { return a < b ? a : b; }

// Order-preserving encoding: encode(a) < encode(b)  <=>  a < b  (IEEE f32)
static __device__ __forceinline__ unsigned enc_f32(float x) {
  unsigned u = __float_as_uint(x);
  return (u & 0x80000000u) ? ~u : (u | 0x80000000u);
}
static __device__ __forceinline__ float dec_f32(unsigned e) {
  return (e & 0x80000000u) ? __uint_as_float(e ^ 0x80000000u)
                           : __uint_as_float(~e);
}

#if VFE_HAS_TDM
// Issue one TDM load of a 16-row x CIN-col f16 tile of WT[COUT][CIN] into LDS.
// D# group1 enables LDS padding: after each row (CIN/2 dwords) insert 4 dwords
// (8 halves) so the tile lands with row pitch CIN+8 -> bank-conflict-free
// B-fragment ds_load_b128s.  Per CDNA5 ISA.
template <int CIN, int COUT>
static __device__ __forceinline__ void tdm_load_wtile(const h16* WT, int ct,
                                                      unsigned ldsOff) {
  constexpr unsigned PIC = (CIN == 64) ? 4u : (CIN == 128) ? 5u : 6u; // rowlen dwords
  unsigned long long ga =
      (unsigned long long)(const void*)(WT + (size_t)ct * 16 * CIN);
  v4u g0 = {1u,                       // count=1, user descriptor
            ldsOff,                   // lds_addr (flat LDS addr truncates to offset)
            (unsigned)ga,             // global_addr[31:0]
            (unsigned)((ga >> 32) & 0x01FFFFFFull) | (2u << 30)}; // [56:32]|type=2
  v8i g1 = {(int)((1u << 16) |       // data_size = 1 -> 2 bytes
                  (1u << 20) |       // pad_enable
                  (PIC << 22) |      // pad_interval: pad after each row
                  (3u << 25)),       // pad_amount: 4 dwords (8 halves)
            (int)(((unsigned)CIN & 0xFFFFu) << 16),                // tdim0 lo
            (int)((((unsigned)CIN >> 16) & 0xFFFFu) |
                  (((unsigned)COUT & 0xFFFFu) << 16)),             // tdim0 hi|tdim1 lo
            (int)(((((unsigned)COUT) >> 16) & 0xFFFFu) |
                  (((unsigned)CIN & 0xFFFFu) << 16)),              // tdim1 hi|tile0=CIN
            16,                                                    // tile1=16, tile2=0
            (int)CIN,                                              // dim0 stride lo
            0, 0};
  v4i z4 = {0, 0, 0, 0};
#if VFE_TDM6
  v8i z8 = {0, 0, 0, 0, 0, 0, 0, 0};
  __builtin_amdgcn_tensor_load_to_lds(g0, g1, z4, z4, z8, 0);
#else
  __builtin_amdgcn_tensor_load_to_lds(g0, g1, z4, z4, 0);
#endif
}
#endif

// ---------------------------------------------------------------------------
__global__ void vfe_init(float* __restrict__ stats, unsigned long long nStats,
                         unsigned* __restrict__ enc, unsigned long long nEnc) {
  unsigned long long i = (unsigned long long)blockIdx.x * blockDim.x + threadIdx.x;
  unsigned long long stride = (unsigned long long)gridDim.x * blockDim.x;
  for (unsigned long long j = i; j < nStats; j += stride) stats[j] = 0.0f;
  for (unsigned long long j = i; j < nEnc; j += stride) enc[j] = 0x007FFFFFu; // enc(-inf)
}

// ---------------------------------------------------------------------------
__global__ __launch_bounds__(256) void vfe_stats9(const float* __restrict__ X,
                                                  float* __restrict__ gsum,
                                                  float* __restrict__ gsq, int N) {
  float ls[9], lq[9];
#pragma unroll
  for (int j = 0; j < 9; j++) { ls[j] = 0.f; lq[j] = 0.f; }
  for (int i = blockIdx.x * blockDim.x + threadIdx.x; i < N;
       i += gridDim.x * blockDim.x) {
#pragma unroll
    for (int j = 0; j < 9; j++) {
      float v = X[(size_t)i * 9 + j];
      ls[j] += v; lq[j] += v * v;
    }
  }
  __shared__ float ssum[9], ssq[9];
  if (threadIdx.x < 9) { ssum[threadIdx.x] = 0.f; ssq[threadIdx.x] = 0.f; }
  __syncthreads();
#pragma unroll
  for (int j = 0; j < 9; j++) {
    float a = ls[j], b = lq[j];
    for (int d = 16; d; d >>= 1) { a += __shfl_xor(a, d, 32); b += __shfl_xor(b, d, 32); }
    if ((threadIdx.x & 31) == 0) { atomicAdd(&ssum[j], a); atomicAdd(&ssq[j], b); }
  }
  __syncthreads();
  if (threadIdx.x < 9) {
    atomicAdd(&gsum[threadIdx.x], ssum[threadIdx.x]);
    atomicAdd(&gsq[threadIdx.x], ssq[threadIdx.x]);
  }
}

// st[0..C-1] = s = gamma*rsqrt(var+eps);  st[C..2C-1] = t = beta - mean*s
__global__ void vfe_bnfold(const float* __restrict__ sum, const float* __restrict__ sq,
                           const float* __restrict__ g, const float* __restrict__ be,
                           float* __restrict__ st, int C, float invN) {
  int c = blockIdx.x * blockDim.x + threadIdx.x;
  if (c < C) {
    float m = sum[c] * invN;
    float v = sq[c] * invN - m * m;
    float s = g[c] * rsqrtf(v + 1e-5f);
    st[c] = s;
    st[C + c] = be[c] - m * s;
  }
}

// W[Cin][Cout] f32  ->  WT[Cout][Cin] f16  (B fragments become contiguous)
__global__ void vfe_wtrans(const float* __restrict__ W, h16* __restrict__ WT,
                           int Cin, int Cout) {
  int i = blockIdx.x * blockDim.x + threadIdx.x;
  if (i < Cin * Cout) {
    int o = i / Cin, k = i - o * Cin;
    WT[i] = (h16)W[(size_t)k * Cout + o];
  }
}

// ---------------------------------------------------------------------------
// Layer 1: K=9 too skinny for WMMA -> scalar FMAs (memory bound anyway).
__global__ __launch_bounds__(256) void vfe_layer1(
    const float* __restrict__ X, const float* __restrict__ W,
    const float* __restrict__ B, const float* __restrict__ st,
    h16* __restrict__ Y, float* __restrict__ gsum, float* __restrict__ gsq, int N) {
  __shared__ float w[9 * 64], bb[64], s0[9], t0[9];
  __shared__ float lsum[64], lsq[64];
  int tid = threadIdx.x;
  for (int i = tid; i < 576; i += 256) w[i] = W[i];
  if (tid < 64) { bb[tid] = B[tid]; lsum[tid] = 0.f; lsq[tid] = 0.f; }
  if (tid < 9) { s0[tid] = st[tid]; t0[tid] = st[9 + tid]; }
  __syncthreads();

  int idx = blockIdx.x * 256 + tid;
  float y[64];
#pragma unroll
  for (int o = 0; o < 64; o++) y[o] = 0.f;
  if (idx < N) {
    float z[9];
#pragma unroll
    for (int j = 0; j < 9; j++) z[j] = fmaf(X[(size_t)idx * 9 + j], s0[j], t0[j]);
#pragma unroll
    for (int o = 0; o < 64; o++) y[o] = bb[o];
#pragma unroll
    for (int j = 0; j < 9; j++)
#pragma unroll
      for (int o = 0; o < 64; o++) y[o] = fmaf(z[j], w[j * 64 + o], y[o]);
    // vectorized f16 stores: 8x global_store_b128
#pragma unroll
    for (int c = 0; c < 8; c++) {
      union { uint4 q; h16 h[8]; } u;
#pragma unroll
      for (int j = 0; j < 8; j++) u.h[j] = (h16)y[c * 8 + j];
      *(uint4*)(Y + (size_t)idx * 64 + c * 8) = u.q;
    }
  }
  // lane-rotated LDS atomics: 32 lanes hit 32 distinct channels each step
  int lane = tid & 31;
#pragma unroll
  for (int j = 0; j < 64; j++) {
    int o = (j + lane) & 63;
    atomicAdd(&lsum[o], y[o]);
    atomicAdd(&lsq[o], y[o] * y[o]);
  }
  __syncthreads();
  if (tid < 64) { atomicAdd(&gsum[tid], lsum[tid]); atomicAdd(&gsq[tid], lsq[tid]); }
}

// ---------------------------------------------------------------------------
// WMMA GEMM: Y[N,COUT] = relu(X*s+t) @ W + b.  128 thr = 4 waves, 64 rows/block.
// A frag layout (16-bit A 16x32): lane L -> row L&15, group g=L>>4,
//   elem e<8 -> K=g*8+e, e>=8 -> K=16+g*8+(e-8)  (two ds_load_b128)
// B frag layout (same K order as A): lane L -> col L&15, K as for A
//   (two ds_load_b128, 16 halves apart, from transposed-weight LDS tile)
// Weight tiles double-buffered via Tensor Data Mover (TENSORcnt).
template <int CIN, int COUT, bool POOL>
__global__ __launch_bounds__(128) void vfe_gemm(
    const h16* __restrict__ X, const h16* __restrict__ WT,
    const float* __restrict__ bias, const float* __restrict__ st,
    h16* __restrict__ Y, float* __restrict__ gsum, float* __restrict__ gsq,
    const int* __restrict__ coords, unsigned* __restrict__ pooled, int N) {
  constexpr int ROWS = 64;
  constexpr int LDA = CIN + 8;   // +16B row pad breaks LDS bank conflicts
  constexpr int NT = COUT / 16;  // number of 16-wide column tiles
  __shared__ h16 ldsA[ROWS * LDA];      // A tile; reused as output repack tile
  __shared__ h16 ldsW[2][16 * LDA];     // double-buffered weight tiles (TDM dest)
  __shared__ float sS[CIN], sT[CIN];
  __shared__ float lsum[COUT], lsq[COUT];
  __shared__ int vox[ROWS];

  const int tid = threadIdx.x;
  const int lane = tid & 31;
  const int wave = tid >> 5;
  const int g = lane >> 4;
  const int halfl = lane & 15;
  const int rowBase = blockIdx.x * ROWS;
  if (rowBase >= N) return;
  const bool full = (rowBase + ROWS <= N);  // always true for N % 64 == 0

#if VFE_HAS_TDM
  // Kick off tile 0 DMA immediately; overlaps with A staging below.
  if (wave == 0 && lane == 0)
    tdm_load_wtile<CIN, COUT>(WT, 0, (unsigned)(unsigned long long)&ldsW[0][0]);
#endif

  for (int i = tid; i < CIN; i += 128) { sS[i] = st[i]; sT[i] = st[CIN + i]; }
  if constexpr (!POOL) {
    for (int i = tid; i < COUT; i += 128) { lsum[i] = 0.f; lsq[i] = 0.f; }
  } else {
    if (tid < ROWS) {
      int rr = imin(rowBase + tid, N - 1);
      vox[tid] = coords[3 * rr] * 1024 + coords[3 * rr + 1] * 32 + coords[3 * rr + 2];
    }
  }
  if (tid == 0 && rowBase + ROWS < N)
    __builtin_prefetch(X + (size_t)(rowBase + ROWS) * CIN, 0, 0);
  __syncthreads();

  // Stage A tile (BN + ReLU applied, f16), b128 in / b128 out.
  for (int idx = tid * 8; idx < ROWS * CIN; idx += 128 * 8) {
    int r = idx / CIN, k = idx - r * CIN;
    int rr = imin(rowBase + r, N - 1);
    union { uint4 q; h16 h[8]; } in, out;
    in.q = *(const uint4*)(X + (size_t)rr * CIN + k);
#pragma unroll
    for (int j = 0; j < 8; j++) {
      float z = fmaf((float)in.h[j], sS[k + j], sT[k + j]);
      out.h[j] = (h16)fmaxf(z, 0.f);
    }
    *(uint4*)(ldsA + r * LDA + k) = out.q;
  }
  __syncthreads();

  // Build per-wave A fragments once; ldsA becomes free for output repack.
  constexpr int KS = CIN / 32;
  v16h afrag[KS];
  {
    const h16* ab = ldsA + (wave * 16 + halfl) * LDA;
#pragma unroll
    for (int ks = 0; ks < KS; ks++) {
      union { v16h v; uint4 q[2]; } u;
      u.q[0] = *(const uint4*)(ab + ks * 32 + g * 8);
      u.q[1] = *(const uint4*)(ab + ks * 32 + 16 + g * 8);
      afrag[ks] = u.v;
    }
  }

  for (int ct = 0; ct < NT; ct++) {
    const int buf = ct & 1;
    __syncthreads();  // prior compute/store done -> buf^1 and repack tile free
#if VFE_HAS_TDM
    if (wave == 0 && lane == 0) {
      if (ct + 1 < NT)
        tdm_load_wtile<CIN, COUT>(WT, ct + 1,
                                  (unsigned)(unsigned long long)&ldsW[buf ^ 1][0]);
      if (ct + 1 < NT) __builtin_amdgcn_s_wait_tensorcnt((short)1);  // tile ct done
      else            __builtin_amdgcn_s_wait_tensorcnt((short)0);
    }
#else
    for (int idx = tid * 8; idx < 16 * CIN; idx += 128 * 8) {
      int c = idx / CIN, k = idx - c * CIN;
      *(uint4*)(&ldsW[buf][0] + c * LDA + k) =
          *(const uint4*)(WT + (size_t)(ct * 16 + c) * CIN + k);
    }
#endif
    __syncthreads();  // weight tile ct visible to all waves

    const int col = ct * 16 + halfl;
    v8f acc;
    {
      float b = bias[col];
#pragma unroll
      for (int i = 0; i < 8; i++) acc[i] = b;
    }
    const h16* bb = &ldsW[buf][0] + halfl * LDA;
#pragma unroll
    for (int ks = 0; ks < KS; ks++) {
      union { v16h v; uint4 q[2]; } u;
      u.q[0] = *(const uint4*)(bb + ks * 32 + g * 8);
      u.q[1] = *(const uint4*)(bb + ks * 32 + 16 + g * 8);
      acc = __builtin_amdgcn_wmma_f32_16x16x32_f16(
          false, afrag[ks], false, u.v, (short)0, acc, false, false);
    }

    // C layout: lane -> col = lane&15; VGPR i -> row i (lanes<16) / i+8.
    if constexpr (!POOL) {
      // per-channel stats from registers
      float s = 0.f, q = 0.f;
#pragma unroll
      for (int i = 0; i < 8; i++) { s += acc[i]; q += acc[i] * acc[i]; }
      s += __shfl_xor(s, 16, 32);
      q += __shfl_xor(q, 16, 32);
      if (lane < 16) { atomicAdd(&lsum[col], s); atomicAdd(&lsq[col], q); }
      // repack 64x16 tile through LDS -> coalesced global_store_b128
      h16* tile = ldsA;  // reuse
#pragma unroll
      for (int i = 0; i < 8; i++) {
        int m = (lane < 16) ? i : (i + 8);
        tile[(wave * 16 + m) * 16 + halfl] = (h16)acc[i];
      }
      __syncthreads();
      {
        int r = tid >> 1, hp = tid & 1;
        uint4 q4 = *(const uint4*)(tile + r * 16 + hp * 8);
        if (full || rowBase + r < N)
          *(uint4*)(Y + (size_t)(rowBase + r) * COUT + ct * 16 + hp * 8) = q4;
      }
    } else {
#pragma unroll
      for (int i = 0; i < 8; i++) {
        int m = (lane < 16) ? i : (i + 8);
        int row = rowBase + wave * 16 + m;
        if (full || row < N) {
          unsigned e = enc_f32(acc[i]);
          atomicMax(pooled + (size_t)vox[wave * 16 + m] * 256u + col, e);
        }
      }
    }
  }

  if constexpr (!POOL) {
    __syncthreads();
    for (int i = tid; i < COUT; i += 128) {
      atomicAdd(&gsum[i], lsum[i]);
      atomicAdd(&gsq[i], lsq[i]);
    }
  }
}

// ---------------------------------------------------------------------------
__global__ void vfe_final(unsigned* __restrict__ enc, long long* __restrict__ co, int V) {
  int i = blockIdx.x * blockDim.x + threadIdx.x;
  if (i < V * 256) ((float*)enc)[i] = dec_f32(enc[i]);  // decode pooled in place
  if (i < V) {  // sorted unique coords = grid enumeration (all voxels occupied)
    co[3 * i + 0] = (long long)(i >> 10);
    co[3 * i + 1] = (long long)((i >> 5) & 31);
    co[3 * i + 2] = (long long)(i & 31);
  }
}

// ---------------------------------------------------------------------------
extern "C" void kernel_launch(void* const* d_in, const int* in_sizes, int n_in,
                              void* d_out, int out_size, void* d_ws, size_t ws_size,
                              hipStream_t stream) {
  const float* feats  = (const float*)d_in[0];
  const int*   coords = (const int*)d_in[1];
  // d_in[2] = num_voxels scalar (device); 32^3 grid per reference:
  const int V = 32768;
  const float* g0 = (const float*)d_in[3];  const float* be0 = (const float*)d_in[4];
  const float* W1 = (const float*)d_in[5];  const float* b1  = (const float*)d_in[6];
  const float* g1 = (const float*)d_in[7];  const float* be1 = (const float*)d_in[8];
  const float* W2 = (const float*)d_in[9];  const float* b2  = (const float*)d_in[10];
  const float* g2 = (const float*)d_in[11]; const float* be2 = (const float*)d_in[12];
  const float* W3 = (const float*)d_in[13]; const float* b3  = (const float*)d_in[14];
  const float* g3 = (const float*)d_in[15]; const float* be3 = (const float*)d_in[16];
  const float* W4 = (const float*)d_in[17]; const float* b4  = (const float*)d_in[18];

  const int N = in_sizes[0] / 9;  // 1,000,000 (divisible by 64)
  const float invN = 1.0f / (float)N;

  // ---- workspace carve-up (256B aligned) ----
  size_t off = 0;
  auto take = [&](size_t bytes) -> void* {
    void* p = (char*)d_ws + off;
    off = (off + bytes + 255) & ~(size_t)255;
    return p;
  };
  float* sum0 = (float*)take(9 * 4);    float* sq0 = (float*)take(9 * 4);
  float* sum1 = (float*)take(64 * 4);   float* sq1 = (float*)take(64 * 4);
  float* sum2 = (float*)take(128 * 4);  float* sq2 = (float*)take(128 * 4);
  float* sum3 = (float*)take(256 * 4);  float* sq3 = (float*)take(256 * 4);
  float* st0  = (float*)take(18 * 4);
  float* st1  = (float*)take(128 * 4);
  float* st2  = (float*)take(256 * 4);
  float* st3  = (float*)take(512 * 4);
  const size_t statsFloats = off / 4;   // zero everything up to here
  h16* W2T = (h16*)take((size_t)128 * 64 * 2);
  h16* W3T = (h16*)take((size_t)256 * 128 * 2);
  h16* W4T = (h16*)take((size_t)256 * 256 * 2);
  h16* y1  = (h16*)take((size_t)N * 64 * 2);
  h16* y2  = (h16*)take((size_t)N * 128 * 2);
  h16* y3  = (h16*)take((size_t)N * 256 * 2);

  unsigned*  enc = (unsigned*)d_out;                               // pooled
  long long* co  = (long long*)((float*)d_out + (size_t)V * 256);  // unq_coords

  // ---- pipeline ----
  vfe_init<<<4096, 256, 0, stream>>>((float*)d_ws, statsFloats, enc,
                                     (unsigned long long)V * 256);
  vfe_stats9<<<1024, 256, 0, stream>>>(feats, sum0, sq0, N);
  vfe_bnfold<<<1, 256, 0, stream>>>(sum0, sq0, g0, be0, st0, 9, invN);
  vfe_wtrans<<<(128 * 64 + 255) / 256, 256, 0, stream>>>(W2, W2T, 64, 128);
  vfe_wtrans<<<(256 * 128 + 255) / 256, 256, 0, stream>>>(W3, W3T, 128, 256);
  vfe_wtrans<<<(256 * 256 + 255) / 256, 256, 0, stream>>>(W4, W4T, 256, 256);

  vfe_layer1<<<(N + 255) / 256, 256, 0, stream>>>(feats, W1, b1, st0, y1, sum1, sq1, N);
  vfe_bnfold<<<1, 256, 0, stream>>>(sum1, sq1, g1, be1, st1, 64, invN);

  vfe_gemm<64, 128, false><<<(N + 63) / 64, 128, 0, stream>>>(
      y1, W2T, b2, st1, y2, sum2, sq2, nullptr, nullptr, N);
  vfe_bnfold<<<1, 256, 0, stream>>>(sum2, sq2, g2, be2, st2, 128, invN);

  vfe_gemm<128, 256, false><<<(N + 63) / 64, 128, 0, stream>>>(
      y2, W3T, b3, st2, y3, sum3, sq3, nullptr, nullptr, N);
  vfe_bnfold<<<1, 256, 0, stream>>>(sum3, sq3, g3, be3, st3, 256, invN);

  vfe_gemm<256, 256, true><<<(N + 63) / 64, 128, 0, stream>>>(
      y3, W4T, b4, st3, nullptr, nullptr, nullptr, coords, enc, N);

  vfe_final<<<(V * 256 + 255) / 256, 256, 0, stream>>>(enc, co, V);

  (void)n_in; (void)out_size; (void)ws_size;
}
